// TTTWrapper_34746285424695
// MI455X (gfx1250) — matrix operand order repaired and checked by reading, not compiled, not run on hardware
//
#include <hip/hip_runtime.h>

#define Bsz   4
#define Tlen  8192
#define DMdim 1024
#define DIdim 2048
#define CHUNK 256
#define NCHUNK (Tlen / CHUNK)
#define LDS_STRIDE 40  // 32 + 8 halves padding -> conflict-free 16-lane column reads

typedef __bf16 bf16_t;
typedef bf16_t bf16x8  __attribute__((ext_vector_type(8)));
typedef bf16_t bf16x16 __attribute__((ext_vector_type(16)));
typedef float  floatx8 __attribute__((ext_vector_type(8)));

// A-matrix 16x32 bf16 fragment: lane(0-15)->M=lane, K runs {half*8..+8, 16+half*8..+8}
__device__ __forceinline__ bf16x16 frag_a(const bf16_t* tile, int row0) {
  const int lane = threadIdx.x & 31;
  const int hf   = lane >> 4;
  const int r    = lane & 15;
  const bf16_t* p = tile + (row0 + r) * LDS_STRIDE;
  bf16x8 lo = *(const bf16x8*)(p + hf * 8);
  bf16x8 hi = *(const bf16x8*)(p + 16 + hf * 8);
  bf16x16 f;
#pragma unroll
  for (int j = 0; j < 8; ++j) { f[j] = lo[j]; f[j + 8] = hi[j]; }
  return f;
}

// B-matrix 32x16 bf16 fragment: lane->N, K runs {half*8..+8, 16+half*8..+8}
// tile stored as [n][k] so per-lane K is contiguous
__device__ __forceinline__ bf16x16 frag_b(const bf16_t* tile, int col0) {
  const int lane = threadIdx.x & 31;
  const int hf   = lane >> 4;
  const int n    = lane & 15;
  const bf16_t* p = tile + (col0 + n) * LDS_STRIDE;
  bf16x8 lo = *(const bf16x8*)(p + hf * 8);
  bf16x8 hi = *(const bf16x8*)(p + 16 + hf * 8);
  bf16x16 f;
#pragma unroll
  for (int j = 0; j < 8; ++j) { f[j] = lo[j]; f[j + 8] = hi[j]; }
  return f;
}

// zero dW (output tail) and per-chunk gradient-norm accumulators
__global__ __launch_bounds__(256) void ttt_init(float* __restrict__ dW,
                                                float* __restrict__ sumsq) {
  const size_t n = (size_t)Bsz * DMdim * DIdim;
  const size_t stride = (size_t)gridDim.x * blockDim.x;
  for (size_t i = (size_t)blockIdx.x * blockDim.x + threadIdx.x; i < n; i += stride)
    dW[i] = 0.0f;
  const size_t g = (size_t)blockIdx.x * blockDim.x + threadIdx.x;
  if (g < (size_t)NCHUNK * Bsz) sumsq[g] = 0.0f;
}

// out[b, chunk*C + c, m] = sum_k z[c,k] * (W0[m,k] + dW[b,m,k]) + bias[m]
__global__ __launch_bounds__(256)
void ttt_gemm_out(const float* __restrict__ z, const float* __restrict__ W0,
                  const float* __restrict__ dW, const float* __restrict__ bias,
                  float* __restrict__ out, int chunk) {
  __shared__ bf16_t Atile[128 * LDS_STRIDE];  // [c][k]
  __shared__ bf16_t Btile[128 * LDS_STRIDE];  // [m][k] (= W_eff rows)

  const int tid  = threadIdx.x;
  const int b    = blockIdx.y;
  const int tn   = blockIdx.x & 7;   // DM / 128
  const int tm   = blockIdx.x >> 3;  // C  / 128
  const int wave = tid >> 5;
  const int wm   = (wave >> 2) * 64;
  const int wn   = (wave & 3) * 32;

  const float* zc    = z  + ((size_t)b * Tlen + (size_t)chunk * CHUNK + (size_t)tm * 128) * DIdim;
  const float* wrow  = W0 + (size_t)tn * 128 * DIdim;
  const float* dwrow = dW + ((size_t)b * DMdim + (size_t)tn * 128) * DIdim;

  floatx8 acc[4][2];
#pragma unroll
  for (int i = 0; i < 4; ++i)
#pragma unroll
    for (int j = 0; j < 2; ++j)
#pragma unroll
      for (int e = 0; e < 8; ++e) acc[i][j][e] = 0.0f;

  const int lrow = tid >> 3;       // 0..31
  const int lk   = (tid & 7) * 4;  // 0..28

  for (int kt = 0; kt < DIdim; kt += 32) {
    __syncthreads();
#pragma unroll
    for (int pass = 0; pass < 4; ++pass) {
      const int row = lrow + pass * 32;
      const float4 zv = *(const float4*)(zc + (size_t)row * DIdim + kt + lk);
      bf16_t* da = &Atile[row * LDS_STRIDE + lk];
      da[0] = (bf16_t)zv.x; da[1] = (bf16_t)zv.y; da[2] = (bf16_t)zv.z; da[3] = (bf16_t)zv.w;
      const float4 wv = *(const float4*)(wrow  + (size_t)row * DIdim + kt + lk);
      const float4 dv = *(const float4*)(dwrow + (size_t)row * DIdim + kt + lk);
      bf16_t* db = &Btile[row * LDS_STRIDE + lk];
      db[0] = (bf16_t)(wv.x + dv.x); db[1] = (bf16_t)(wv.y + dv.y);
      db[2] = (bf16_t)(wv.z + dv.z); db[3] = (bf16_t)(wv.w + dv.w);
    }
    __syncthreads();

    bf16x16 af[4], bfr[2];
#pragma unroll
    for (int i = 0; i < 4; ++i) af[i] = frag_a(Atile, wm + i * 16);
#pragma unroll
    for (int j = 0; j < 2; ++j) bfr[j] = frag_b(Btile, wn + j * 16);
#pragma unroll
    for (int i = 0; i < 4; ++i)
#pragma unroll
      for (int j = 0; j < 2; ++j)
        acc[i][j] = __builtin_amdgcn_wmma_f32_16x16x32_bf16(
            false, af[i], false, bfr[j], (short)0, acc[i][j], false, false);
  }

  const int lane = tid & 31;
  const int hf   = lane >> 4;
  const int col  = lane & 15;
  float* outc = out + ((size_t)b * Tlen + (size_t)chunk * CHUNK + (size_t)tm * 128) * DMdim;
#pragma unroll
  for (int i = 0; i < 4; ++i)
#pragma unroll
    for (int j = 0; j < 2; ++j) {
      const int gn = tn * 128 + wn + j * 16 + col;
      const float bv = bias[gn];
#pragma unroll
      for (int r = 0; r < 8; ++r) {
        const int grow = wm + i * 16 + hf * 8 + r;
        outc[(size_t)grow * DMdim + gn] = acc[i][j][r] + bv;
      }
    }
}

// G_raw[b,m,d] = sum_c vhat[c,m] * z[c,d]; also accumulate per-batch Frobenius^2
__global__ __launch_bounds__(256)
void ttt_gemm_g(const float* __restrict__ vhat, const float* __restrict__ z,
                float* __restrict__ G, float* __restrict__ sumsq, int chunk) {
  __shared__ bf16_t Atile[128 * LDS_STRIDE];  // [m][c]
  __shared__ bf16_t Btile[128 * LDS_STRIDE];  // [d][c]
  __shared__ float  lred;

  const int tid  = threadIdx.x;
  const int b    = blockIdx.y;
  const int tn   = blockIdx.x & 15;  // DI / 128
  const int tm   = blockIdx.x >> 4;  // DM / 128
  const int wave = tid >> 5;
  const int wm   = (wave >> 2) * 64;
  const int wn   = (wave & 3) * 32;

  const float* vc = vhat + ((size_t)b * Tlen + (size_t)chunk * CHUNK) * DMdim + tm * 128;
  const float* zc = z    + ((size_t)b * Tlen + (size_t)chunk * CHUNK) * DIdim + tn * 128;

  floatx8 acc[4][2];
#pragma unroll
  for (int i = 0; i < 4; ++i)
#pragma unroll
    for (int j = 0; j < 2; ++j)
#pragma unroll
      for (int e = 0; e < 8; ++e) acc[i][j][e] = 0.0f;

  if (tid == 0) lred = 0.0f;

  const int lc = tid >> 5;         // 0..7
  const int lm = (tid & 31) * 4;   // 0..124

  for (int kt = 0; kt < CHUNK; kt += 32) {
    __syncthreads();
#pragma unroll
    for (int pass = 0; pass < 4; ++pass) {
      const int c = lc + pass * 8;  // 0..31
      const float4 av = *(const float4*)(vc + (size_t)(kt + c) * DMdim + lm);
      Atile[(lm + 0) * LDS_STRIDE + c] = (bf16_t)av.x;
      Atile[(lm + 1) * LDS_STRIDE + c] = (bf16_t)av.y;
      Atile[(lm + 2) * LDS_STRIDE + c] = (bf16_t)av.z;
      Atile[(lm + 3) * LDS_STRIDE + c] = (bf16_t)av.w;
      const float4 bv = *(const float4*)(zc + (size_t)(kt + c) * DIdim + lm);
      Btile[(lm + 0) * LDS_STRIDE + c] = (bf16_t)bv.x;
      Btile[(lm + 1) * LDS_STRIDE + c] = (bf16_t)bv.y;
      Btile[(lm + 2) * LDS_STRIDE + c] = (bf16_t)bv.z;
      Btile[(lm + 3) * LDS_STRIDE + c] = (bf16_t)bv.w;
    }
    __syncthreads();

    bf16x16 af[4], bfr[2];
#pragma unroll
    for (int i = 0; i < 4; ++i) af[i] = frag_a(Atile, wm + i * 16);
#pragma unroll
    for (int j = 0; j < 2; ++j) bfr[j] = frag_b(Btile, wn + j * 16);
#pragma unroll
    for (int i = 0; i < 4; ++i)
#pragma unroll
      for (int j = 0; j < 2; ++j)
        acc[i][j] = __builtin_amdgcn_wmma_f32_16x16x32_bf16(
            false, af[i], false, bfr[j], (short)0, acc[i][j], false, false);
  }

  const int lane = tid & 31;
  const int hf   = lane >> 4;
  const int col  = lane & 15;
  float local = 0.0f;
  float* Gc = G + ((size_t)b * DMdim + (size_t)tm * 128) * DIdim + tn * 128;
#pragma unroll
  for (int i = 0; i < 4; ++i)
#pragma unroll
    for (int j = 0; j < 2; ++j)
#pragma unroll
      for (int r = 0; r < 8; ++r) {
        const float v = acc[i][j][r];
        local += v * v;
        const int gm = wm + i * 16 + hf * 8 + r;
        const int gd = wn + j * 16 + col;
        Gc[(size_t)gm * DIdim + gd] = v;
      }

  atomicAdd(&lred, local);
  __syncthreads();
  if (tid == 0) atomicAdd(&sumsq[chunk * Bsz + b], lred);
}

// dW = decay*dW + (eta/C)*G_raw * min(tau/(||(eta/C)G||_F + eps), 1)
__global__ __launch_bounds__(256)
void ttt_update(const float* __restrict__ G, float* __restrict__ dW,
                const float* __restrict__ sumsq,
                const float* __restrict__ log_lr, const float* __restrict__ log_decay,
                int chunk) {
  const float eta    = expf(log_lr[0]);
  const float decay  = 1.0f / (1.0f + expf(-log_decay[0]));
  const float factor = eta / (float)CHUNK;
  const size_t q      = (size_t)DMdim * DIdim / 4;  // float4s per batch
  const size_t total4 = (size_t)Bsz * q;
  const size_t stride = (size_t)gridDim.x * blockDim.x;
  for (size_t idx = (size_t)blockIdx.x * blockDim.x + threadIdx.x; idx < total4; idx += stride) {
    const int b = (int)(idx / q);
    const float s = sumsq[chunk * Bsz + b];
    const float gnorm = factor * sqrtf(s);
    const float scale = fminf(1.0f / (gnorm + 1e-8f), 1.0f) * factor;
    float4 g = ((const float4*)G)[idx];
    float4 w = ((float4*)dW)[idx];
    w.x = decay * w.x + g.x * scale;
    w.y = decay * w.y + g.y * scale;
    w.z = decay * w.z + g.z * scale;
    w.w = decay * w.w + g.w * scale;
    ((float4*)dW)[idx] = w;
  }
}

extern "C" void kernel_launch(void* const* d_in, const int* in_sizes, int n_in,
                              void* d_out, int out_size, void* d_ws, size_t ws_size,
                              hipStream_t stream) {
  const float* z         = (const float*)d_in[0];
  const float* vhat      = (const float*)d_in[1];
  const float* W0        = (const float*)d_in[2];
  const float* bias      = (const float*)d_in[3];
  const float* log_lr    = (const float*)d_in[4];
  const float* log_decay = (const float*)d_in[5];

  float* out = (float*)d_out;
  float* dW  = out + (size_t)Bsz * Tlen * DMdim;    // second output: running state
  float* G     = (float*)d_ws;                      // 32 MB raw gradient buffer
  float* sumsq = G + (size_t)Bsz * DMdim * DIdim;   // [NCHUNK][B]

  ttt_init<<<2048, 256, 0, stream>>>(dW, sumsq);
  for (int c = 0; c < NCHUNK; ++c) {
    ttt_gemm_out<<<dim3(16, Bsz), 256, 0, stream>>>(z, W0, dW, bias, out, c);
    ttt_gemm_g<<<dim3(128, Bsz), 256, 0, stream>>>(vhat, z, G, sumsq, c);
    ttt_update<<<dim3(1024), 256, 0, stream>>>(G, dW, sumsq, log_lr, log_decay, c);
  }
}
